// maskPredictor_41704132444943
// MI455X (gfx1250) — matrix operand order repaired and checked by reading, not compiled, not run on hardware
//
#include <hip/hip_runtime.h>
#include <hip/hip_bf16.h>
#include <math.h>

// ---------- types for WMMA fragments ----------
typedef __attribute__((ext_vector_type(16))) __bf16 v16bf;
typedef __attribute__((ext_vector_type(4)))  __bf16 v4bf;
typedef __attribute__((ext_vector_type(8)))  float  v8f;
typedef __attribute__((ext_vector_type(4)))  float  v4f;
typedef __attribute__((ext_vector_type(4)))  unsigned int u32x4;

union Frag {
    v16bf v;
    u32x4 q[2];
};

#define DIM 768
#define NB  16
#define TQ  1024
#define TK  1024
#define EPSN 1e-8f

__device__ __forceinline__ float bf2f(unsigned short h) {
    return __uint_as_float(((unsigned int)h) << 16);
}

// ---------- kernel 0: f32 -> bf16 conversion (vectorized, native cvt) ----------
__global__ __launch_bounds__(256)
void cvt_f32_bf16_kernel(const float* __restrict__ src, __bf16* __restrict__ dst, int n4) {
    int i = blockIdx.x * 256 + threadIdx.x;
    if (i < n4) {
        v4f x = *(const v4f*)(src + (size_t)i * 4);
        *(v4bf*)(dst + (size_t)i * 4) = __builtin_convertvector(x, v4bf);
    }
}

// ---------- kernel 1/2: projection  Out[m,n] = sum_d X[m,d]*W[n,d] + bias[n]  (bf16 out) ----------
// grid: (M/128, DIM/64), block: 256 (8 waves). Wave w owns rows w*16..w*16+15, all share 64 cols.
__global__ __launch_bounds__(256)
void proj_kernel(const float* __restrict__ X,        // [M, 768] f32
                 const __bf16* __restrict__ Wb,      // [768, 768] bf16 (row n, fast d)
                 const float* __restrict__ bias,     // [768]
                 __bf16* __restrict__ Out)           // [M, 768] bf16
{
    __shared__ __align__(16) __bf16 sA[128 * 32];    // 8 KB staging, f32->bf16

    const int tid  = threadIdx.x;
    const int wave = tid >> 5;
    const int lane = tid & 31;
    const int l16  = lane & 15;
    const int hh   = lane >> 4;
    const int mblock = blockIdx.x * 128;
    const int nbase  = blockIdx.y * 64;

    v8f acc[4];
    for (int t = 0; t < 4; ++t)
        for (int j = 0; j < 8; ++j) acc[t][j] = 0.0f;

    for (int kk = 0; kk < DIM / 32; ++kk) {
        const int kbase = kk * 32;
        // cooperative stage: 128 rows x 32 cols, f32 -> bf16 via native packed cvt
        #pragma unroll
        for (int i = 0; i < 4; ++i) {
            int e0 = (tid + 256 * i) * 4;            // 0..4092, float4 never crosses a row
            int r = e0 >> 5, c = e0 & 31;
            v4f x = *(const v4f*)(X + (size_t)(mblock + r) * DIM + kbase + c);
            *(v4bf*)(sA + e0) = __builtin_convertvector(x, v4bf);
        }
        __syncthreads();

        // A fragment: row = wave*16 + l16, contiguous K-run [16h, 16h+16)
        Frag a;
        const __bf16* ap = &sA[(wave * 16 + l16) * 32 + 16 * hh];
        a.q[0] = *(const u32x4*)(ap);
        a.q[1] = *(const u32x4*)(ap + 8);

        #pragma unroll
        for (int t = 0; t < 4; ++t) {
            // B fragment: col n = nbase + t*16 + l16, contiguous K-run [16h, 16h+16)
            Frag b;
            const __bf16* bp = &Wb[(size_t)(nbase + t * 16 + l16) * DIM + kbase + 16 * hh];
            b.q[0] = *(const u32x4*)(bp);
            b.q[1] = *(const u32x4*)(bp + 8);
            acc[t] = __builtin_amdgcn_wmma_f32_16x16x32_bf16(
                false, a.v, false, b.v, (short)0, acc[t], false, false);
        }
        __syncthreads();
    }

    // epilogue: bias add, bf16 store (native cvt)
    #pragma unroll
    for (int t = 0; t < 4; ++t) {
        int n = nbase + t * 16 + l16;
        float bv = bias[n];
        #pragma unroll
        for (int j = 0; j < 8; ++j) {
            int m = mblock + wave * 16 + j + 8 * hh;
            Out[(size_t)m * DIM + n] = (__bf16)(acc[t][j] + bv);
        }
    }
}

// ---------- kernel 3: per-row inverse L2 norm of a bf16 matrix ----------
// one wave per row; grid = rows/8, block 256
__global__ __launch_bounds__(256)
void rownorm_kernel(const __bf16* __restrict__ Xb, float* __restrict__ inv) {
    int row  = blockIdx.x * 8 + (threadIdx.x >> 5);
    int lane = threadIdx.x & 31;
    const unsigned int* p = (const unsigned int*)(Xb + (size_t)row * DIM);  // 384 dwords
    float s = 0.0f;
    for (int i = lane; i < DIM / 2; i += 32) {
        unsigned int u = p[i];
        float a = bf2f((unsigned short)(u & 0xFFFFu));
        float b = bf2f((unsigned short)(u >> 16));
        s += a * a + b * b;
    }
    for (int m = 16; m >= 1; m >>= 1) s += __shfl_xor(s, m, 32);
    if (lane == 0) inv[row] = 1.0f / fmaxf(sqrtf(s), EPSN);
}

// ---------- kernel 4: fused cosine-scores + masked max + MLP + sigmoid ----------
// grid = B*8 blocks, block 256 (8 waves). Wave w: 16 q-rows, k-chunks of 64 (4 B-tiles / A-load).
__global__ __launch_bounds__(256)
void scores_kernel(const __bf16* __restrict__ Qb,      // [B*Tq, 768] bf16
                   const __bf16* __restrict__ Kb,      // [B*Tk, 768] bf16
                   const float* __restrict__ qinv,     // [B*Tq]
                   const float* __restrict__ kinv,     // [B*Tk]
                   const unsigned char* __restrict__ pad, // [B*Tk] bool
                   const float* __restrict__ W1p,      // [16]
                   const float* __restrict__ b1p,      // [16]
                   const float* __restrict__ W2p,      // [16]
                   const float* __restrict__ b2p,      // [1]
                   float* __restrict__ out)            // [B*Tq]
{
    const int tid  = threadIdx.x;
    const int wave = tid >> 5;
    const int lane = tid & 31;
    const int l16  = lane & 15;
    const int hh   = lane >> 4;
    const int bb   = blockIdx.x >> 3;            // batch
    const int qt   = blockIdx.x & 7;             // 128-row macro tile within batch
    const int qrow0 = bb * TQ + qt * 128 + wave * 16;

    // preload per-row inverse norms for the 8 accumulator rows of this lane
    float qiv[8];
    #pragma unroll
    for (int j = 0; j < 8; ++j) qiv[j] = qinv[qrow0 + j + 8 * hh];

    const __bf16* aBase = Qb + (size_t)(qrow0 + l16) * DIM;

    float vmax[8];
    #pragma unroll
    for (int j = 0; j < 8; ++j) vmax[j] = -__builtin_inff();

    for (int kc = 0; kc < TK / 64; ++kc) {
        const int krow0 = bb * TK + kc * 64;     // global k row of first tile

        const __bf16* kbp[4];
        #pragma unroll
        for (int t = 0; t < 4; ++t)
            kbp[t] = Kb + (size_t)(krow0 + t * 16 + l16) * DIM;

        // prefetch next chunk's rows into cache while this chunk computes
        if (kc + 1 < TK / 64) {
            #pragma unroll
            for (int t = 0; t < 4; ++t)
                __builtin_prefetch(kbp[t] + (size_t)64 * DIM, 0, 0);
        }

        v8f acc[4];
        #pragma unroll
        for (int t = 0; t < 4; ++t)
            for (int j = 0; j < 8; ++j) acc[t][j] = 0.0f;

        #pragma unroll 2
        for (int dd = 0; dd < DIM / 32; ++dd) {
            const int kb = dd * 32;
            Frag a;
            a.q[0] = *(const u32x4*)(aBase + kb + 16 * hh);
            a.q[1] = *(const u32x4*)(aBase + kb + 16 * hh + 8);
            #pragma unroll
            for (int t = 0; t < 4; ++t) {
                Frag f;
                f.q[0] = *(const u32x4*)(kbp[t] + kb + 16 * hh);
                f.q[1] = *(const u32x4*)(kbp[t] + kb + 16 * hh + 8);
                acc[t] = __builtin_amdgcn_wmma_f32_16x16x32_bf16(
                    false, a.v, false, f.v, (short)0, acc[t], false, false);
            }
        }

        // epilogue: scale to cosine, mask, running max
        const float ninf = -__builtin_inff();
        #pragma unroll
        for (int t = 0; t < 4; ++t) {
            float ki = kinv[krow0 + t * 16 + l16];
            bool  pd = pad[krow0 + t * 16 + l16] != 0;
            #pragma unroll
            for (int j = 0; j < 8; ++j) {
                float s = pd ? ninf : acc[t][j] * qiv[j] * ki;
                vmax[j] = fmaxf(vmax[j], s);
            }
        }
    }

    // reduce max across the 16 lanes (columns) of each half-wave
    #pragma unroll
    for (int j = 0; j < 8; ++j) {
        float v = vmax[j];
        v = fmaxf(v, __shfl_xor(v, 8, 32));
        v = fmaxf(v, __shfl_xor(v, 4, 32));
        v = fmaxf(v, __shfl_xor(v, 2, 32));
        v = fmaxf(v, __shfl_xor(v, 1, 32));
        vmax[j] = v;
    }

    __shared__ float smax[8][16];
    if (l16 == 0) {
        #pragma unroll
        for (int j = 0; j < 8; ++j) smax[wave][j + 8 * hh] = vmax[j];
    }
    __syncthreads();

    // tiny MLP 1->16->1 + sigmoid, one lane per q-row
    if (lane < 16) {
        float x = smax[wave][lane];
        float o = b2p[0];
        #pragma unroll
        for (int i = 0; i < 16; ++i) {
            float h = fmaxf(x * W1p[i] + b1p[i], 0.0f);
            o += h * W2p[i];
        }
        o = 1.0f / (1.0f + __expf(-o));
        out[qrow0 + lane] = o;
    }
}

// ---------- host launch ----------
extern "C" void kernel_launch(void* const* d_in, const int* in_sizes, int n_in,
                              void* d_out, int out_size, void* d_ws, size_t ws_size,
                              hipStream_t stream) {
    const float* image = (const float*)d_in[0];              // [16,32,32,768] == [16384,768]
    const float* text  = (const float*)d_in[1];              // [16,1024,768]  == [16384,768]
    const unsigned char* pad = (const unsigned char*)d_in[2];// [16,32,32] bool == [16384]
    const float* Wq = (const float*)d_in[3];
    const float* bq = (const float*)d_in[4];
    const float* Wk = (const float*)d_in[5];
    const float* bk = (const float*)d_in[6];
    const float* W1 = (const float*)d_in[7];                 // [16,1]
    const float* b1 = (const float*)d_in[8];                 // [16]
    const float* W2 = (const float*)d_in[9];                 // [1,16]
    const float* b2 = (const float*)d_in[10];                // [1]
    float* out = (float*)d_out;                              // [16,1024]

    const size_t M = (size_t)NB * TQ;                        // 16384 rows each side
    // workspace carving (all offsets 256B aligned)
    char* ws = (char*)d_ws;
    __bf16* qb  = (__bf16*)(ws);                             // 25165824 B
    __bf16* kb  = (__bf16*)(ws + 25165824);                  // 25165824 B
    __bf16* wqb = (__bf16*)(ws + 50331648);                  // 1179648 B
    __bf16* wkb = (__bf16*)(ws + 51511296);                  // 1179648 B
    float* qinv = (float*)(ws + 52690944);                   // 65536 B
    float* kinv = (float*)(ws + 52756480);                   // 65536 B

    // 1) weights f32 -> bf16
    {
        int n4 = DIM * DIM / 4;
        int g = (n4 + 255) / 256;
        cvt_f32_bf16_kernel<<<g, 256, 0, stream>>>(Wq, wqb, n4);
        cvt_f32_bf16_kernel<<<g, 256, 0, stream>>>(Wk, wkb, n4);
    }
    // 2) projections via WMMA
    {
        dim3 grid((unsigned)(M / 128), DIM / 64);
        proj_kernel<<<grid, 256, 0, stream>>>(text,  wqb, bq, qb);
        proj_kernel<<<grid, 256, 0, stream>>>(image, wkb, bk, kb);
    }
    // 3) inverse row norms
    {
        int g = (int)(M / 8);
        rownorm_kernel<<<g, 256, 0, stream>>>(qb, qinv);
        rownorm_kernel<<<g, 256, 0, stream>>>(kb, kinv);
    }
    // 4) fused cosine scores + masked max + MLP
    {
        scores_kernel<<<NB * 8, 256, 0, stream>>>(qb, kb, qinv, kinv, pad,
                                                  W1, b1, W2, b2, out);
    }
}
